// PAN_DGAL_3633542332836
// MI455X (gfx1250) — matrix operand order repaired and checked by reading, not compiled, not run on hardware
//
#include <hip/hip_runtime.h>
#include <hip/hip_bf16.h>

typedef __attribute__((ext_vector_type(16))) __bf16 v16bf;
typedef __attribute__((ext_vector_type(8)))  __bf16 v8bf;
typedef __attribute__((ext_vector_type(8)))  float  v8f;

namespace {
constexpr int   kB  = 2;
constexpr int   kC  = 32;
constexpr int   kIC = 4;
constexpr int   kH  = 128;
constexpr int   kW  = 128;
constexpr int   kLH = 32;
constexpr int   kLW = 32;
constexpr int   kNP = 1024;     // 32*32 coarse positions
constexpr int   kNQ = 16384;    // 128*128 fine positions
constexpr int   kF  = 196;      // 4*7*7 patch features
constexpr int   kFP = 208;      // padded to 13*16
constexpr int   kG  = 49;
constexpr int   kGP = 64;       // padded to 2*32 (WMMA K)
constexpr float kScale = 10.0f;
}

// ---------------------------------------------------------------- precompute
// b1 = conv3x3(b, w_g)+b_g (pad 1), b2 = conv1x1(b, w_theta)+b_theta
__global__ void k_conv_pre(const float* __restrict__ b,
                           const float* __restrict__ w_g, const float* __restrict__ b_g,
                           const float* __restrict__ w_t, const float* __restrict__ b_t,
                           float* __restrict__ b1, float* __restrict__ b2) {
  int idx = blockIdx.x * blockDim.x + threadIdx.x;
  if (idx >= kB * kIC * kH * kW) return;
  int w  = idx % kW;
  int h  = (idx / kW) % kH;
  int oc = (idx / (kW * kH)) % kIC;
  int nb = idx / (kW * kH * kIC);
  const float* bp = b + (size_t)nb * kC * kH * kW;
  float acc1 = b_g[oc];
  float acc2 = b_t[oc];
  for (int c = 0; c < kC; ++c) {
    const float* bc = bp + (size_t)c * kH * kW;
    acc2 += bc[h * kW + w] * w_t[oc * kC + c];
    for (int ki = 0; ki < 3; ++ki) {
      int ih = h - 1 + ki;
      if (ih < 0 || ih >= kH) continue;
      for (int kj = 0; kj < 3; ++kj) {
        int iw = w - 1 + kj;
        if (iw < 0 || iw >= kW) continue;
        acc1 += bc[ih * kW + iw] * w_g[(oc * kC + c) * 9 + ki * 3 + kj];
      }
    }
  }
  b1[idx] = acc1;
  b2[idx] = acc2;
}

// thr/bias: 7x7 conv, stride 4, pad ((1,2),(1,2)) -> (B,1024)
__global__ void k_thrbias(const float* __restrict__ b,
                          const float* __restrict__ w_thr, const float* __restrict__ b_thr,
                          const float* __restrict__ w_bias, const float* __restrict__ b_bias,
                          float* __restrict__ thr, float* __restrict__ bias) {
  int idx = blockIdx.x * blockDim.x + threadIdx.x;
  if (idx >= kB * kNP) return;
  int pw = idx % kLW;
  int ph = (idx / kLW) % kLH;
  int nb = idx / kNP;
  float a1 = b_thr[0], a2 = b_bias[0];
  for (int c = 0; c < kC; ++c) {
    const float* bc = b + ((size_t)nb * kC + c) * kH * kW;
    for (int ki = 0; ki < 7; ++ki) {
      int ih = ph * 4 - 1 + ki;
      if (ih < 0 || ih >= kH) continue;
      for (int kj = 0; kj < 7; ++kj) {
        int iw = pw * 4 - 1 + kj;
        if (iw < 0 || iw >= kW) continue;
        float v = bc[ih * kW + iw];
        a1 += v * w_thr [c * 49 + ki * 7 + kj];
        a2 += v * w_bias[c * 49 + ki * 7 + kj];
      }
    }
  }
  thr[idx]  = a1;
  bias[idx] = a2;
}

// wi = relu(p28(b1) @ w_fc1^T + b_fc1) -> bf16 [B][1024][64] (g padded)
__global__ void k_wi(const float* __restrict__ b1, const float* __restrict__ w_fc1,
                     const float* __restrict__ b_fc1, __bf16* __restrict__ wi) {
  int idx = blockIdx.x * blockDim.x + threadIdx.x;
  if (idx >= kB * kNP * kGP) return;
  int g  = idx % kGP;
  int p  = (idx / kGP) % kNP;
  int nb = idx / (kGP * kNP);
  if (g >= kG) { wi[idx] = (__bf16)0.0f; return; }
  int pw = p % kLW, ph = p / kLW;
  float acc = b_fc1[g];
  for (int c = 0; c < kIC; ++c) {
    const float* bc = b1 + ((size_t)nb * kIC + c) * kH * kW;
    for (int ki = 0; ki < 7; ++ki) {
      int ih = ph * 4 - 1 + ki;
      for (int kj = 0; kj < 7; ++kj) {
        int iw = pw * 4 - 1 + kj;
        float v = (ih >= 0 && ih < kH && iw >= 0 && iw < kW) ? bc[ih * kW + iw] : 0.0f;
        acc += v * w_fc1[g * kF + c * 49 + ki * 7 + kj];
      }
    }
  }
  wi[idx] = (__bf16)fmaxf(acc, 0.0f);
}

// xi = relu(p112(b1) @ w_fc2^T + b_fc2) -> bf16 [B][16384][64]
__global__ void k_xi(const float* __restrict__ b1, const float* __restrict__ w_fc2,
                     const float* __restrict__ b_fc2, __bf16* __restrict__ xi) {
  int idx = blockIdx.x * blockDim.x + threadIdx.x;
  if (idx >= kB * kNQ * kGP) return;
  int g  = idx % kGP;
  int q  = (idx / kGP) % kNQ;
  int nb = idx / (kGP * kNQ);
  if (g >= kG) { xi[idx] = (__bf16)0.0f; return; }
  int qw = q % kW, qh = q / kW;
  float acc = b_fc2[g];
  for (int c = 0; c < kIC; ++c) {
    const float* bc = b1 + ((size_t)nb * kIC + c) * kH * kW;
    for (int ki = 0; ki < 7; ++ki) {
      int ih = qh - 3 + ki;
      for (int kj = 0; kj < 7; ++kj) {
        int iw = qw - 3 + kj;
        float v = (ih >= 0 && ih < kH && iw >= 0 && iw < kW) ? bc[ih * kW + iw] : 0.0f;
        acc += v * w_fc2[g * kF + c * 49 + ki * 7 + kj];
      }
    }
  }
  xi[idx] = (__bf16)fmaxf(acc, 0.0f);
}

// xisum[nb][g] = sum_q xi[nb][q][g]  (f32, from the same bf16 data the WMMAs see)
// -> replaces an entire WMMA score pass: mean_p = wi_p . xisum / NQ
__global__ void k_xisum(const __bf16* __restrict__ xi, float* __restrict__ xisum) {
  __shared__ float part[4][kGP];
  int nb = blockIdx.x;                 // one block per batch
  int g  = threadIdx.x & 63;
  int sl = threadIdx.x >> 6;           // 4 q-slices
  float acc = 0.0f;
  const __bf16* base = xi + (size_t)nb * kNQ * kGP + g;
  for (int q = sl * (kNQ / 4); q < (sl + 1) * (kNQ / 4); ++q)
    acc += (float)base[(size_t)q * kGP];
  part[sl][g] = acc;
  __syncthreads();
  if (threadIdx.x < kGP)
    xisum[nb * kGP + threadIdx.x] = part[0][threadIdx.x] + part[1][threadIdx.x] +
                                    part[2][threadIdx.x] + part[3][threadIdx.x];
}

// p112t: bf16 transposed patches of b2: [B][208][16384]  (f-major, q contiguous)
__global__ void k_p112t(const float* __restrict__ b2, __bf16* __restrict__ p112t) {
  int idx = blockIdx.x * blockDim.x + threadIdx.x;
  if (idx >= kB * kFP * kNQ) return;
  int q  = idx % kNQ;
  int f  = (idx / kNQ) % kFP;
  int nb = idx / (kNQ * kFP);
  float v = 0.0f;
  if (f < kF) {
    int c = f / 49, r = f % 49, ki = r / 7, kj = r % 7;
    int qh = q / kW, qw = q % kW;
    int ih = qh - 3 + ki, iw = qw - 3 + kj;
    if (ih >= 0 && ih < kH && iw >= 0 && iw < kW)
      v = b2[((size_t)nb * kIC + c) * kH * kW + ih * kW + iw];
  }
  p112t[idx] = (__bf16)v;
}

// ---------------------------------------------------------------- WMMA helpers
__device__ __forceinline__ v16bf load_b16(const __bf16* p) {
  v8bf x0 = *(const v8bf*)p;
  v8bf x1 = *(const v8bf*)(p + 16);
  v16bf r;
#pragma unroll
  for (int i = 0; i < 8; ++i) { r[i] = x0[i]; r[i + 8] = x1[i]; }
  return r;
}

__device__ __forceinline__ v8f score_tile(v16bf a0, v16bf a1, const __bf16* xb) {
  // B fragment (32x16, K=g): this lane's K-halves are contiguous in xi row
  v16bf b0 = load_b16(xb);        // K  0..31 window
  v16bf b1 = load_b16(xb + 32);   // K 32..63 window
  v8f c = {};
  c = __builtin_amdgcn_wmma_f32_16x16x32_bf16(false, a0, false, b0, (short)0, c, false, false);
  c = __builtin_amdgcn_wmma_f32_16x16x32_bf16(false, a1, false, b1, (short)0, c, false, false);
  return c;
}

// ---------------------------------------------------------------- fused attention
__global__ __launch_bounds__(128) void k_attn(
    const __bf16* __restrict__ wiB, const __bf16* __restrict__ xiB,
    const __bf16* __restrict__ p112t, const float* __restrict__ thr,
    const float* __restrict__ bias, const float* __restrict__ xisum,
    float* __restrict__ attn_out) {
  __shared__ float s_m[4][16];
  __shared__ float s_c[4][16];
  __shared__ float s_mean[16];
  __shared__ float s_M[16];
  __shared__ float s_Linv[16];
  __shared__ __align__(16) __bf16 s_aw[4][16][32];   // per-wave A staging
  __shared__ float s_opart[4][16 * kFP];             // per-wave output partials

  const int tid  = threadIdx.x;
  const int wave = tid >> 5;
  const int lane = tid & 31;
  const int ln   = lane & 15;
  const int hi   = lane >> 4;        // 0: rows 0-7 / K-lo, 1: rows 8-15 / K-hi
  const int off  = hi * 8;
  const int nb   = blockIdx.x >> 6;
  const int p0   = (blockIdx.x & 63) * 16;

  // ---- analytic row means: mean_p = wi_p . xisum / NQ  (replaces a WMMA pass)
  if (tid < 16) {
    const __bf16* wr = wiB + ((size_t)nb * kNP + p0 + tid) * kGP;
    const float* xs = xisum + nb * kGP;
    float acc = 0.0f;
    for (int g = 0; g < kGP; ++g) acc += (float)wr[g] * xs[g];
    s_mean[tid] = acc / (float)kNQ;
  }

  // ---- A fragments for wi tile (16 rows x 64 K), constant across all passes
  const __bf16* wrow = wiB + ((size_t)nb * kNP + p0 + ln) * kGP;
  v16bf a0, a1;
  {
    v8bf l0 = *(const v8bf*)(wrow + 0  + off);
    v8bf h0 = *(const v8bf*)(wrow + 16 + off);
    v8bf l1 = *(const v8bf*)(wrow + 32 + off);
    v8bf h1 = *(const v8bf*)(wrow + 48 + off);
#pragma unroll
    for (int i = 0; i < 8; ++i) {
      a0[i] = l0[i]; a0[i + 8] = h0[i];
      a1[i] = l1[i]; a1[i + 8] = h1[i];
    }
  }
  const __bf16* xibase = xiB + ((size_t)nb * kNQ + ln) * kGP + hi * 8;
  __syncthreads();

  float mn[8], th[8], bi[8];
#pragma unroll
  for (int r = 0; r < 8; ++r) {
    int m = hi * 8 + r;
    mn[r] = s_mean[m];
    th[r] = thr [(size_t)nb * kNP + p0 + m];
    bi[r] = bias[(size_t)nb * kNP + p0 + m];
  }

  // ================= pass 1: global max of masked logits (no exps) =================
  float mrow[8];
#pragma unroll
  for (int r = 0; r < 8; ++r) mrow[r] = -1e30f;
  for (int ch = wave; ch < kNQ / 16; ch += 4) {
    v8f s = score_tile(a0, a1, xibase + (size_t)(ch * 16) * kGP);
#pragma unroll
    for (int r = 0; r < 8; ++r) {
      float sc  = s[r];
      float msk = fmaxf(sc - mn[r] * th[r] + bi[r], 0.0f);
      mrow[r]   = fmaxf(mrow[r], sc * msk * kScale);   // masked-out -> t = 0
    }
  }
#pragma unroll
  for (int xm = 1; xm < 16; xm <<= 1)
#pragma unroll
    for (int r = 0; r < 8; ++r) mrow[r] = fmaxf(mrow[r], __shfl_xor(mrow[r], xm, 32));
  if (ln == 0)
#pragma unroll
    for (int r = 0; r < 8; ++r) s_m[wave][hi * 8 + r] = mrow[r];
  __syncthreads();
  if (tid < 16)
    s_M[tid] = fmaxf(fmaxf(s_m[0][tid], s_m[1][tid]), fmaxf(s_m[2][tid], s_m[3][tid]));
  __syncthreads();
  float Mr[8];
#pragma unroll
  for (int r = 0; r < 8; ++r) Mr[r] = s_M[hi * 8 + r];

  // ===== pass 2: numerators -> output GEMM; denominator accumulated in-flight =====
  v8f o[13];
#pragma unroll
  for (int nf = 0; nf < 13; ++nf) { v8f z = {}; o[nf] = z; }
  float sw[8], cn[8];                      // sum of live exps, count of masked
#pragma unroll
  for (int r = 0; r < 8; ++r) { sw[r] = 0.0f; cn[r] = 0.0f; }
  const __bf16* pbase = p112t + ((size_t)nb * kFP + ln) * kNQ + hi * 8;

  for (int ch = wave; ch < kNQ / 32; ch += 4) {
    int q0 = ch * 32;
    // two 16x16 score tiles -> unnormalized softmax weights staged via LDS
#pragma unroll
    for (int j = 0; j < 2; ++j) {
      v8f s = score_tile(a0, a1, xibase + (size_t)(q0 + j * 16) * kGP);
#pragma unroll
      for (int r = 0; r < 8; ++r) {
        float sc  = s[r];
        float msk = fmaxf(sc - mn[r] * th[r] + bi[r], 0.0f);
        float t   = sc * msk * kScale;
        bool  live = (msk > 0.0f);
        float e   = live ? __expf(t - Mr[r]) : 0.0f;   // numerator * mask_b
        sw[r] += e;
        cn[r] += live ? 0.0f : 1.0f;
        s_aw[wave][hi * 8 + r][j * 16 + ln] = (__bf16)e;
      }
    }
    asm volatile("s_wait_dscnt 0" ::: "memory");   // wave-local LDS transpose RAW
    v16bf aw;
    {
      const __bf16* row = &s_aw[wave][ln][0];
      v8bf x0 = *(const v8bf*)(row + off);
      v8bf x1 = *(const v8bf*)(row + 16 + off);
#pragma unroll
      for (int i = 0; i < 8; ++i) { aw[i] = x0[i]; aw[i + 8] = x1[i]; }
    }
    __builtin_prefetch((const void*)(xibase + (size_t)(q0 + 128) * kGP), 0, 1);
#pragma unroll
    for (int nf = 0; nf < 13; ++nf) {
      v16bf bv = load_b16(pbase + (size_t)nf * 16 * kNQ + q0);
      o[nf] = __builtin_amdgcn_wmma_f32_16x16x32_bf16(false, aw, false, bv, (short)0,
                                                      o[nf], false, false);
    }
  }

  // reduce denominator pieces: lanes (xor shuffles) then waves (LDS)
#pragma unroll
  for (int xm = 1; xm < 16; xm <<= 1)
#pragma unroll
    for (int r = 0; r < 8; ++r) {
      sw[r] += __shfl_xor(sw[r], xm, 32);
      cn[r] += __shfl_xor(cn[r], xm, 32);
    }
  if (ln == 0)
#pragma unroll
    for (int r = 0; r < 8; ++r) { s_m[wave][hi * 8 + r] = sw[r]; s_c[wave][hi * 8 + r] = cn[r]; }
  __syncthreads();
  if (tid < 16) {
    float S = s_m[0][tid] + s_m[1][tid] + s_m[2][tid] + s_m[3][tid];
    float C = s_c[0][tid] + s_c[1][tid] + s_c[2][tid] + s_c[3][tid];
    // masked entries contribute exp(0 - M) each to the softmax denominator
    float L = S + ((C > 0.0f) ? C * __expf(-s_M[tid]) : 0.0f);
    s_Linv[tid] = 1.0f / L;
  }

  // combine the 4 wave partials through LDS; apply 1/L per row here
#pragma unroll
  for (int nf = 0; nf < 13; ++nf)
#pragma unroll
    for (int r = 0; r < 8; ++r)
      s_opart[wave][(hi * 8 + r) * kFP + nf * 16 + ln] = o[nf][r];
  __syncthreads();
  for (int e = tid; e < 16 * kFP; e += 128) {
    float v = s_opart[0][e] + s_opart[1][e] + s_opart[2][e] + s_opart[3][e];
    int m = e / kFP, f = e % kFP;
    attn_out[((size_t)nb * kNP + p0 + m) * kFP + f] = v * s_Linv[m];
  }
}

// ---------------------------------------------------------------- fold + count-normalize
__global__ void k_fold(const float* __restrict__ attn_out, float* __restrict__ out) {
  int idx = blockIdx.x * blockDim.x + threadIdx.x;
  if (idx >= kB * kIC * kH * kW) return;
  int w  = idx % kW;
  int h  = (idx / kW) % kH;
  int c  = (idx / (kW * kH)) % kIC;
  int nb = idx / (kW * kH * kIC);
  int hp = h + 3, wp = w + 3;
  float acc = 0.0f;
  int cnt = 0;
  for (int ki = 0; ki < 7; ++ki) {
    int r = hp - ki;
    if (r < 0 || (r & 3) || (r >> 2) >= kLH) continue;
    int ph = r >> 2;
    for (int kj = 0; kj < 7; ++kj) {
      int s = wp - kj;
      if (s < 0 || (s & 3) || (s >> 2) >= kLW) continue;
      int pw = s >> 2;
      acc += attn_out[((size_t)nb * kNP + ph * kLW + pw) * kFP + c * 49 + ki * 7 + kj];
      ++cnt;
    }
  }
  out[idx] = acc / (float)cnt;
}

// ---------------------------------------------------------------- launcher
extern "C" void kernel_launch(void* const* d_in, const int* in_sizes, int n_in,
                              void* d_out, int out_size, void* d_ws, size_t ws_size,
                              hipStream_t stream) {
  const float* b      = (const float*)d_in[0];
  const float* w_g    = (const float*)d_in[1];
  const float* b_g    = (const float*)d_in[2];
  const float* w_th   = (const float*)d_in[3];
  const float* b_th   = (const float*)d_in[4];
  const float* w_fc1  = (const float*)d_in[5];
  const float* b_fc1  = (const float*)d_in[6];
  const float* w_fc2  = (const float*)d_in[7];
  const float* b_fc2  = (const float*)d_in[8];
  const float* w_thr  = (const float*)d_in[9];
  const float* b_thr  = (const float*)d_in[10];
  const float* w_bias = (const float*)d_in[11];
  const float* b_bias = (const float*)d_in[12];
  float* out = (float*)d_out;

  char* ws = (char*)d_ws;
  float*  b1       = (float*) (ws + 0);          //   512 KB
  float*  b2       = (float*) (ws + 524288);     //   512 KB
  float*  thr      = (float*) (ws + 1048576);    //     8 KB
  float*  bias     = (float*) (ws + 1056768);    //     8 KB
  __bf16* wiB      = (__bf16*)(ws + 1064960);    //   256 KB
  __bf16* xiB      = (__bf16*)(ws + 1327104);    //     4 MB
  __bf16* p112t    = (__bf16*)(ws + 5521408);    //    13 MB
  float*  attn_out = (float*) (ws + 19152896);   //   1.7 MB
  float*  xisum    = (float*) (ws + 20856832);   //   512 B    (total ~20.9 MB)

  const int BT = 256;
  int n;

  n = kB * kIC * kH * kW;
  k_conv_pre<<<(n + BT - 1) / BT, BT, 0, stream>>>(b, w_g, b_g, w_th, b_th, b1, b2);

  n = kB * kNP;
  k_thrbias<<<(n + BT - 1) / BT, BT, 0, stream>>>(b, w_thr, b_thr, w_bias, b_bias, thr, bias);

  n = kB * kNP * kGP;
  k_wi<<<(n + BT - 1) / BT, BT, 0, stream>>>(b1, w_fc1, b_fc1, wiB);

  n = kB * kNQ * kGP;
  k_xi<<<(n + BT - 1) / BT, BT, 0, stream>>>(b1, w_fc2, b_fc2, xiB);

  k_xisum<<<kB, 256, 0, stream>>>(xiB, xisum);

  n = kB * kFP * kNQ;
  k_p112t<<<(n + BT - 1) / BT, BT, 0, stream>>>(b2, p112t);

  k_attn<<<kB * (kNP / 16), 128, 0, stream>>>(wiB, xiB, p112t, thr, bias, xisum, attn_out);

  n = kB * kIC * kH * kW;
  k_fold<<<(n + BT - 1) / BT, BT, 0, stream>>>(attn_out, out);
}
